// KernelAggregation_49280454754759
// MI455X (gfx1250) — matrix operand order repaired and checked by reading, not compiled, not run on hardware
//
#include <hip/hip_runtime.h>

typedef __attribute__((ext_vector_type(16))) _Float16 v16h;
typedef __attribute__((ext_vector_type(8)))  _Float16 v8h;
typedef __attribute__((ext_vector_type(8)))  float    v8f;
typedef __attribute__((ext_vector_type(4)))  unsigned int u32x4;
typedef __attribute__((ext_vector_type(4)))  int          i32x4;
typedef __attribute__((ext_vector_type(8)))  int          i32x8;

#define B_SZ 8
#define O_SZ 2048
#define Q_SZ 2048
#define D_SZ 128
#define QT   128          // query rows per block
#define KC   32           // K chunk (one f16 WMMA K)
#define NT   8            // 128/16 column tiles
#define SB_STRIDE 40      // halfs per column row in staged emb (padded, 80B)
#define SW_STRIDE 136     // halfs per row in staged W (padded, 272B)
#define SQ_STRIDE 136     // halfs per row in staged query_emb

#if defined(__has_builtin)
#  if __has_builtin(__builtin_amdgcn_tensor_load_to_lds) && __has_builtin(__builtin_amdgcn_s_wait_tensorcnt)
#    define USE_TDM 1
#  endif
#endif
#ifndef USE_TDM
#  define USE_TDM 0
#endif

#if USE_TDM
// LDS byte offset of a __shared__ object (flat LDS address low bits == wave-relative offset)
__device__ __forceinline__ unsigned lds_off(const void* p) {
  return (unsigned)(unsigned long long)(uintptr_t)p;
}

// Issue one TDM load: 32 rows x 128 f32 (16KB), row-major, global -> LDS.
__device__ __forceinline__ void tdm_load_chunk(const float* gptr, unsigned lds_addr) {
  unsigned long long ga = (unsigned long long)(uintptr_t)gptr;
  // D# group 0: count=1 | lds_addr | global_addr[56:0] | type=2 ("image")
  u32x4 g0 = { 1u,
               lds_addr,
               (unsigned)(ga & 0xffffffffu),
               (unsigned)((ga >> 32) & 0x01ffffffu) | (2u << 30) };
  // D# group 1: data_size=2 (4B); tensor_dim0=128; tensor_dim1=2048;
  //             tile_dim0=128, tile_dim1=32; tensor_dim0_stride=128
  i32x8 g1 = { (int)(2u << 16),          // [17:16] data_size
               (int)(128u << 16),        // [79:48] tensor_dim0 = 128 (low half)
               (int)(2048u << 16),       // [111:80] tensor_dim1 = 2048 (low half)
               (int)(128u << 16),        // [127:112] tile_dim0 = 128
               (int)32,                  // [143:128] tile_dim1 = 32
               (int)128,                 // [207:160] tensor_dim0_stride = 128
               0, 0 };
  i32x4 z4 = { 0, 0, 0, 0 };             // groups 2/3 unused (2D tile)
#if __clang_major__ >= 23
  i32x8 z8 = { 0, 0, 0, 0, 0, 0, 0, 0 };
  __builtin_amdgcn_tensor_load_to_lds(g0, g1, z4, z4, z8, 0);
#else
  __builtin_amdgcn_tensor_load_to_lds(g0, g1, z4, z4, 0);
#endif
}
#endif // USE_TDM

__global__ __launch_bounds__(256)
void rbf_agg_proj_kernel(const float* __restrict__ obs_emb,
                         const float* __restrict__ obs_times,
                         const float* __restrict__ query_times,
                         const float* __restrict__ obs_mask,
                         const float* __restrict__ log_sigma,
                         const float* __restrict__ W_proj,
                         const float* __restrict__ b_proj,
                         float* __restrict__ out)
{
  // s_u: f16 transposed emb chunk (GEMM1, 10240B) then normalized query_emb (GEMM2, 34816B)
  __shared__ __align__(16) _Float16 s_u[QT * SQ_STRIDE];
  __shared__ __align__(16) _Float16 s_W[D_SZ * SW_STRIDE];
#if USE_TDM
  __shared__ __align__(16) float s_raw[2][KC * D_SZ];   // TDM ping-pong, 2 x 16KB
#endif
  __shared__ float s_ot[KC];
  __shared__ float s_om[KC];
  __shared__ float s_bias[D_SZ];

  _Float16* s_emb = s_u;
  _Float16* s_QE  = s_u;

  const int tid  = threadIdx.x;
  const int wv   = tid >> 5;        // wave 0..7
  const int lane = tid & 31;
  const int m    = lane & 15;       // row within 16x16 tile (A) / column (B)
  const int hi   = lane >> 4;       // half-wave selector

  const int b  = blockIdx.x >> 4;          // batch
  const int q0 = (blockIdx.x & 15) * QT;   // query tile origin

  const float* embB = obs_emb   + (size_t)b * O_SZ * D_SZ;
  const float* otB  = obs_times + (size_t)b * O_SZ;
  const float* omB  = obs_mask  + (size_t)b * O_SZ;

#if USE_TDM
  if (wv == 0) tdm_load_chunk(embB, lds_off(&s_raw[0][0]));   // prime chunk 0
#endif

  // ---- stage W_proj (f16, padded rows) + bias, once per block ----
  {
    const float4* W4 = (const float4*)W_proj;
    for (int i = tid; i < (D_SZ * D_SZ) / 4; i += 256) {
      float4 w = W4[i];
      int e = (i * 4) >> 7;
      int d = (i * 4) & (D_SZ - 1);
      _Float16* p = &s_W[e * SW_STRIDE + d];
      p[0] = (_Float16)w.x; p[1] = (_Float16)w.y;
      p[2] = (_Float16)w.z; p[3] = (_Float16)w.w;
    }
    if (tid < D_SZ) s_bias[tid] = b_proj[tid];
  }

  const float inv_sigma = __expf(-log_sigma[0]);
  const float nh2 = -0.5f * inv_sigma * inv_sigma;

  const float tq = query_times[(size_t)b * Q_SZ + q0 + wv * 16 + m];

  const v8f zero = {0.f,0.f,0.f,0.f,0.f,0.f,0.f,0.f};
  v8f acc[NT];
  #pragma unroll
  for (int n = 0; n < NT; n++) acc[n] = zero;
  float wsum = 0.f;

  // ================= GEMM1: weights(QTxO) @ emb(OxD), unnormalized =========
  for (int it = 0; it < O_SZ / KC; ++it) {
    const int o0 = it * KC;
#if USE_TDM
    if (wv == 0) {
      if (o0 + KC < O_SZ) {       // stream next chunk while current computes
        tdm_load_chunk(embB + (size_t)(o0 + KC) * D_SZ, lds_off(&s_raw[(it + 1) & 1][0]));
        __builtin_amdgcn_s_wait_tensorcnt(1);   // chunk `it` complete (in-order)
      } else {
        __builtin_amdgcn_s_wait_tensorcnt(0);
      }
    }
#endif
    __syncthreads();   // raw chunk visible; prev compute done with s_emb/s_ot
    if (tid < KC)           s_ot[tid]      = otB[o0 + tid];
    else if (tid < 2 * KC)  s_om[tid - KC] = omB[o0 + tid - KC];
    {
      // transpose + f32->f16: s_emb[n][k]; thread t -> n = t&127, k half = t>>7
      int n  = tid & (D_SZ - 1);
      int k8 = tid >> 7;
      v8h t0, t1;
#if USE_TDM
      const float* raw = &s_raw[it & 1][0];
      #pragma unroll
      for (int i = 0; i < 8; i++)
        t0[i] = (_Float16)raw[(k8 * 16 + i) * D_SZ + n];
      #pragma unroll
      for (int i = 0; i < 8; i++)
        t1[i] = (_Float16)raw[(k8 * 16 + 8 + i) * D_SZ + n];
#else
      #pragma unroll
      for (int i = 0; i < 8; i++)
        t0[i] = (_Float16)embB[(size_t)(o0 + k8 * 16 + i) * D_SZ + n];
      #pragma unroll
      for (int i = 0; i < 8; i++)
        t1[i] = (_Float16)embB[(size_t)(o0 + k8 * 16 + 8 + i) * D_SZ + n];
      if (o0 + KC < O_SZ && tid < 128)
        __builtin_prefetch(embB + (size_t)(o0 + KC) * D_SZ + tid * 32, 0, 1);
#endif
      v8h* dst = (v8h*)&s_emb[n * SB_STRIDE + k8 * 16];
      dst[0] = t0; dst[1] = t1;
    }
    __syncthreads();

    // A fragment: RBF weights, this code's contiguous k order: elem e -> K=e+16*hi
    v16h afrag;
    float psum = 0.f;
    #pragma unroll
    for (int e = 0; e < 16; e++) {
      int K = e + hi * 16;
      float td = tq - s_ot[K];
      float w  = __expf(nh2 * td * td) * s_om[K];
      psum += w;
      afrag[e] = (_Float16)w;
    }
    wsum += psum;

    #pragma unroll
    for (int nt = 0; nt < NT; nt++) {
      // B fragment: lane col n=nt*16+m, K = 16*hi + e (contiguous in staged LDS)
      const v8h* p = (const v8h*)&s_emb[(nt * 16 + m) * SB_STRIDE + hi * 16];
      v8h lo = p[0], hh = p[1];
      v16h bfrag;
      #pragma unroll
      for (int e = 0; e < 8; e++) { bfrag[e] = lo[e]; bfrag[8 + e] = hh[e]; }
      acc[nt] = __builtin_amdgcn_wmma_f32_16x16x32_f16(
          false, afrag, false, bfrag, (short)0, acc[nt], false, false);
    }
  }

  // ---- fused row normalization: lane m holds K 0..15, lane m+16 K 16..31
  float s = wsum + __shfl_xor(wsum, 16, 32);
  s = fmaxf(s, 1e-8f);
  const float invS = 1.0f / s;

  __syncthreads();  // all waves done with s_emb before aliasing as s_QE
  #pragma unroll
  for (int r = 0; r < 8; r++) {
    float is = __shfl(invS, r + 8 * hi, 32);        // C row M = r + 8*hi
    int row = wv * 16 + r + 8 * hi;
    #pragma unroll
    for (int nt = 0; nt < NT; nt++)
      s_QE[row * SQ_STRIDE + nt * 16 + m] = (_Float16)(acc[nt][r] * is);
  }
  __syncthreads();

  // ================= GEMM2: query_emb(QTx128) @ W^T(128x128) ===============
  v8f acc2[NT];
  #pragma unroll
  for (int n = 0; n < NT; n++) acc2[n] = zero;

  #pragma unroll
  for (int kc = 0; kc < D_SZ / KC; kc++) {
    // A2: row = wv*16+m, K = kc*32 + 16*hi + e
    const v8h* pa = (const v8h*)&s_QE[(wv * 16 + m) * SQ_STRIDE + kc * 32 + 16 * hi];
    v8h a0 = pa[0], a1 = pa[1];
    v16h a2;
    #pragma unroll
    for (int e = 0; e < 8; e++) { a2[e] = a0[e]; a2[8 + e] = a1[e]; }
    #pragma unroll
    for (int nt = 0; nt < NT; nt++) {
      // B2[k][n] = W[n][k]: lane col n=nt*16+m, K = kc*32 + 16*hi + e
      const v8h* pb = (const v8h*)&s_W[(nt * 16 + m) * SW_STRIDE + kc * 32 + 16 * hi];
      v8h b0 = pb[0], b1 = pb[1];
      v16h b2;
      #pragma unroll
      for (int e = 0; e < 8; e++) { b2[e] = b0[e]; b2[8 + e] = b1[e]; }
      acc2[nt] = __builtin_amdgcn_wmma_f32_16x16x32_f16(
          false, a2, false, b2, (short)0, acc2[nt], false, false);
    }
  }

  // ---- epilogue: + bias, store f32 ----
  float* outB = out + (size_t)b * Q_SZ * D_SZ + (size_t)q0 * D_SZ;
  #pragma unroll
  for (int r = 0; r < 8; r++) {
    int row = wv * 16 + r + 8 * hi;
    #pragma unroll
    for (int nt = 0; nt < NT; nt++) {
      int col = nt * 16 + m;
      outB[(size_t)row * D_SZ + col] = acc2[nt][r] + s_bias[col];
    }
  }
}

extern "C" void kernel_launch(void* const* d_in, const int* in_sizes, int n_in,
                              void* d_out, int out_size, void* d_ws, size_t ws_size,
                              hipStream_t stream) {
  const float* obs_emb     = (const float*)d_in[0];
  const float* obs_times   = (const float*)d_in[1];
  const float* query_times = (const float*)d_in[2];
  const float* obs_mask    = (const float*)d_in[3];
  const float* log_sigma   = (const float*)d_in[4];
  const float* W_proj      = (const float*)d_in[5];
  const float* b_proj      = (const float*)d_in[6];
  float* out = (float*)d_out;

  dim3 grid(B_SZ * (Q_SZ / QT));   // 8 batches * 16 query tiles = 128 blocks
  dim3 block(256);                 // 8 wave32s
  rbf_agg_proj_kernel<<<grid, block, 0, stream>>>(
      obs_emb, obs_times, query_times, obs_mask, log_sigma, W_proj, b_proj, out);
}
